// FBPNet_20134806684416
// MI455X (gfx1250) — compile-verified, matrix operand order checked by reading, not run on hardware
//
#include <hip/hip_runtime.h>

// ---------------------------------------------------------------------------
// Filtered backprojection (FBP) for gfx1250 / MI455X, wave32.
//   1) Ram-Lak filter as circulant GEMM via v_wmma_f32_16x16x32_f16
//   2) LDS-staged backprojection sweep (30-angle chunks, 60KB LDS/block)
//   3) 3x3 gaussian blur + clip
// ---------------------------------------------------------------------------

typedef __attribute__((ext_vector_type(8)))  _Float16 v8h;
typedef __attribute__((ext_vector_type(16))) _Float16 v16h;
typedef __attribute__((ext_vector_type(8)))  float    v8f;

#define BATCH 8
#define NANG  180
#define NDET  512
#define IMG   512
#define HW    (IMG * IMG)
#define NROW  (BATCH * NANG)          // 1440 rows in the filter GEMM

// workspace byte offsets (all 256-aligned where it matters)
#define OFF_H     0u                              // 512 f32 filter taps
#define OFF_CTAB  2048u                           // 180 f32 cos/sqrt2
#define OFF_STAB  2768u                           // 180 f32 sin/sqrt2
#define OFF_XH    4096u                           // 1440*512 f16     (1,474,560 B)
#define OFF_HT    (4096u + 1474560u)              // 512*512  f16     (  524,288 B)
#define OFF_FILT  2002944u                        // 1440*512 f32     (2,949,120 B)
#define OFF_FBP   4952064u                        // 8*512*512 f32    (8,388,608 B)
// total ~13.3 MB

// ---------------------------------------------------------------------------
// 1) Filter impulse response h[n] = irfft(|rfftfreq(512)|)[n]  and cos/sin tables
//    h[n] = (1/512) * ( 2*sum_{k=1}^{255} (k/512) cos(2πkn/512) + 0.5*(-1)^n )
// ---------------------------------------------------------------------------
__global__ void k_init_tables(float* __restrict__ h,
                              float* __restrict__ ctab,
                              float* __restrict__ stab) {
    int t = blockIdx.x * blockDim.x + threadIdx.x;
    if (t < NDET) {
        float n   = (float)t;
        float acc = 0.5f * ((t & 1) ? -1.0f : 1.0f);   // k = D/2 term
        const float w = 6.283185307179586f / 512.0f;
        for (int k = 1; k < 256; ++k)
            acc += 2.0f * ((float)k * (1.0f / 512.0f)) * __cosf(w * (float)k * n);
        h[t] = acc * (1.0f / 512.0f);
    } else if (t < NDET + NANG) {
        int a = t - NDET;
        float ang = (float)a * (3.14159265358979323846f / 180.0f);
        ctab[a] = __cosf(ang) * 0.7071067811865476f;   // fold 1/sqrt(2) in
        stab[a] = __sinf(ang) * 0.7071067811865476f;
    }
}

// ---------------------------------------------------------------------------
// 2) Pack sinogram to f16, build transposed circulant HT[n][k] = h[(n-k)&511]
// ---------------------------------------------------------------------------
__global__ void k_cast_x(const float* __restrict__ x, _Float16* __restrict__ xh, int n) {
    int i = blockIdx.x * blockDim.x + threadIdx.x;
    if (i < n) xh[i] = (_Float16)x[i];
}

__global__ void k_build_ht(const float* __restrict__ h, _Float16* __restrict__ ht) {
    int i = blockIdx.x * blockDim.x + threadIdx.x;
    if (i < NDET * NDET) {
        int n = i >> 9, k = i & 511;
        ht[i] = (_Float16)h[(n - k) & 511];
    }
}

// ---------------------------------------------------------------------------
// 3) Filter GEMM: filt[1440,512] = Xh[1440,512] x H[512,512], fp32 accumulate.
//    One wave -> one 16x16 tile; 8 waves/block -> 16x128 tile; K loop of 16
//    v_wmma_f32_16x16x32_f16. Fragment layouts per CDNA5 ISA 7.12.2 (wave32):
//      A: lane m = lane%16, K in {koff..koff+7} u {koff+16..koff+23}, koff=8*(lane>>4)
//      B: mirrored with N = lane%16 (HT stored row-major by N so both are
//         two contiguous 16B loads per lane)
//      C/D: c[r] -> row r0 + r + 8*(lane>>4), col n0 + lane%16
// ---------------------------------------------------------------------------
__global__ void __launch_bounds__(256)
k_filter_gemm(const _Float16* __restrict__ xh,
              const _Float16* __restrict__ ht,
              float* __restrict__ filt) {
    const int lane = threadIdx.x & 31;
    const int wave = threadIdx.x >> 5;
    const int n0   = blockIdx.x * 128 + wave * 16;
    const int r0   = blockIdx.y * 16;
    const int half = lane >> 4;            // 0 or 1
    const int koff = half * 8;
    const int m    = r0 + (lane & 15);
    const int n    = n0 + (lane & 15);

    const _Float16* arow = xh + (size_t)m * NDET;
    const _Float16* brow = ht + (size_t)n * NDET;

    v8f c = {};
    #pragma unroll 4
    for (int kk = 0; kk < NDET; kk += 32) {
        v8h alo = *(const v8h*)(arow + kk + koff);
        v8h ahi = *(const v8h*)(arow + kk + koff + 16);
        v8h blo = *(const v8h*)(brow + kk + koff);
        v8h bhi = *(const v8h*)(brow + kk + koff + 16);
        v16h af = __builtin_shufflevector(alo, ahi, 0,1,2,3,4,5,6,7,8,9,10,11,12,13,14,15);
        v16h bf = __builtin_shufflevector(blo, bhi, 0,1,2,3,4,5,6,7,8,9,10,11,12,13,14,15);
        // (neg_a, A, neg_b, B, c_mod, C, reuse_a, reuse_b)
        c = __builtin_amdgcn_wmma_f32_16x16x32_f16(false, af, false, bf,
                                                   (short)0, c, false, false);
    }

    const int rowbase = r0 + half * 8;
    #pragma unroll
    for (int r = 0; r < 8; ++r)
        filt[(size_t)(rowbase + r) * NDET + (n0 + (lane & 15))] = c[r];
}

// ---------------------------------------------------------------------------
// 4) Backprojection: per (batch, pixel) accumulate over 180 angles with
//    linear interpolation. Projection rows staged in LDS in 30-angle chunks
//    (30*512*4 = 60KB); cos/sin tables also in LDS. Next chunk prefetched.
// ---------------------------------------------------------------------------
#define ACHUNK 30
__global__ void __launch_bounds__(256)
k_backproject(const float* __restrict__ filt,
              const float* __restrict__ ctab,
              const float* __restrict__ stab,
              float* __restrict__ fbp) {
    __shared__ float sproj[ACHUNK * NDET];
    __shared__ float sc[NANG];
    __shared__ float ss[NANG];

    const int tid = threadIdx.x;
    const int b   = blockIdx.y;
    const int pix = blockIdx.x * 256 + tid;

    for (int i = tid; i < NANG; i += 256) { sc[i] = ctab[i]; ss[i] = stab[i]; }

    const int   px = pix & (IMG - 1);
    const int   py = pix >> 9;
    const float xf = -1.0f + 2.0f * (float)px * (1.0f / 511.0f);
    const float yf = -1.0f + 2.0f * (float)py * (1.0f / 511.0f);

    const float* src = filt + (size_t)b * NANG * NDET;
    float acc = 0.0f;

    for (int a0 = 0; a0 < NANG; a0 += ACHUNK) {
        __syncthreads();                       // tables ready / prev chunk consumed
        for (int i = tid; i < ACHUNK * NDET; i += 256)
            sproj[i] = src[a0 * NDET + i];     // chunk is contiguous in memory
        if (a0 + ACHUNK < NANG)                // hint next chunk toward L2/L0
            __builtin_prefetch(src + (a0 + ACHUNK) * NDET + tid * 4, 0, 1);
        __syncthreads();

        #pragma unroll 5
        for (int a = 0; a < ACHUNK; ++a) {
            float t = xf * sc[a0 + a] + yf * ss[a0 + a];
            float u = (t + 1.0f) * 255.5f;
            u = fminf(fmaxf(u, 0.0f), 511.0f);
            int   i0 = (int)u;                 // u >= 0 -> trunc == floor
            float w  = u - (float)i0;
            int   i1 = (i0 < 511) ? i0 + 1 : 511;
            float p0 = sproj[a * NDET + i0];
            float p1 = sproj[a * NDET + i1];
            acc += p0 + w * (p1 - p0);
        }
    }

    const float scale = (2.0f / 512.0f) * (3.14159265358979f / 180.0f);
    fbp[(size_t)b * HW + pix] = acc * scale;
}

// ---------------------------------------------------------------------------
// 5) 3x3 gaussian blur (sigma=0.5, zero 'SAME' padding) + clip to [-1,1]
// ---------------------------------------------------------------------------
__global__ void __launch_bounds__(256)
k_blur_clip(const float* __restrict__ fbp, float* __restrict__ out) {
    const int b   = blockIdx.y;
    const int pix = blockIdx.x * 256 + threadIdx.x;
    const int px  = pix & (IMG - 1);
    const int py  = pix >> 9;

    // k1 = normalize([e^-2, 1, e^-2])
    const float w0 = 0.10650697891920077f;
    const float w1 = 0.7869860421615985f;
    const float wt[3] = { w0, w1, w0 };

    const float* img = fbp + (size_t)b * HW;
    float s = 0.0f;
    #pragma unroll
    for (int dy = -1; dy <= 1; ++dy) {
        int y = py + dy;
        if (y < 0 || y > IMG - 1) continue;
        #pragma unroll
        for (int dx = -1; dx <= 1; ++dx) {
            int x = px + dx;
            if (x < 0 || x > IMG - 1) continue;
            s += wt[dy + 1] * wt[dx + 1] * img[y * IMG + x];
        }
    }
    out[(size_t)b * HW + pix] = fminf(1.0f, fmaxf(-1.0f, s));
}

// ---------------------------------------------------------------------------
extern "C" void kernel_launch(void* const* d_in, const int* in_sizes, int n_in,
                              void* d_out, int out_size, void* d_ws, size_t ws_size,
                              hipStream_t stream) {
    const float* x   = (const float*)d_in[0];   // [B, A, D] fp32 sinogram
    float*       out = (float*)d_out;           // [B, 1, IMG, IMG] fp32

    char* ws = (char*)d_ws;
    float*    h    = (float*)(ws + OFF_H);
    float*    ctab = (float*)(ws + OFF_CTAB);
    float*    stab = (float*)(ws + OFF_STAB);
    _Float16* xh   = (_Float16*)(ws + OFF_XH);
    _Float16* ht   = (_Float16*)(ws + OFF_HT);
    float*    filt = (float*)(ws + OFF_FILT);
    float*    fbp  = (float*)(ws + OFF_FBP);

    k_init_tables<<<3, 256, 0, stream>>>(h, ctab, stab);
    k_cast_x<<<(NROW * NDET + 255) / 256, 256, 0, stream>>>(x, xh, NROW * NDET);
    k_build_ht<<<(NDET * NDET + 255) / 256, 256, 0, stream>>>(h, ht);
    k_filter_gemm<<<dim3(NDET / 128, NROW / 16), 256, 0, stream>>>(xh, ht, filt);
    k_backproject<<<dim3(HW / 256, BATCH), 256, 0, stream>>>(filt, ctab, stab, fbp);
    k_blur_clip<<<dim3(HW / 256, BATCH), 256, 0, stream>>>(fbp, out);
}
